// MultiHeadAttention_65644280152639
// MI455X (gfx1250) — matrix operand order repaired and checked by reading, not compiled, not run on hardware
//
#include <hip/hip_runtime.h>
#include <hip/hip_bf16.h>
#include <stdint.h>

#define B_  2
#define S_  4096
#define D_  512
#define H_  8
#define DR_ 64

typedef __attribute__((ext_vector_type(16))) _Float16 v16h;
typedef __attribute__((ext_vector_type(8)))  float    v8f;
typedef __attribute__((ext_vector_type(8)))  uint32_t v8u;

// ---- WMMA fragment loaders (per CDNA5 ISA 7.12.2 layouts) --------------------

// A-matrix 16x32 f16: lane&15 = row M, lane>>4 selects K-half (0/8),
// VGPR i packs K = {base+2i, base+2i+1} (pairs contiguous in memory, ld = row stride).
static __device__ __forceinline__ v16h load_afrag(const _Float16* p, int ld) {
  int lane = threadIdx.x & 31;
  int m  = lane & 15;
  int kb = (lane >> 4) << 3;           // 0 or 8
  const _Float16* row = p + (size_t)m * ld;
  v8u u;
#pragma unroll
  for (int i = 0; i < 8; ++i) {
    int k = ((i & 4) ? 16 : 0) + kb + ((i & 3) << 1);
    u[i] = *(const uint32_t*)(row + k);
  }
  return __builtin_bit_cast(v16h, u);
}

// B-matrix 32x16 f16: lane&15 = column N, lane>>4 selects K-half (0/8),
// VGPR v packs K = {base+2v, base+2v+1} (same order as A). Source layout: B[kk,n] at p[n*ld + kk] (K contiguous).
static __device__ __forceinline__ v16h load_bfrag(const _Float16* p, int ld) {
  int lane = threadIdx.x & 31;
  int n  = lane & 15;
  int kb = (lane >> 4) << 3;           // 0 or 8
  const _Float16* col = p + (size_t)n * ld;
  v8u u;
#pragma unroll
  for (int v = 0; v < 8; ++v) u[v] = *(const uint32_t*)(col + ((v & 4) ? 16 : 0) + kb + ((v & 3) << 1));
  return __builtin_bit_cast(v16h, u);
}

static __device__ __forceinline__ v8f wmma_f16(v16h a, v16h b, v8f c) {
  return __builtin_amdgcn_wmma_f32_16x16x32_f16(false, a, false, b, (short)0, c, false, false);
}

// ---- Prep kernels ------------------------------------------------------------

__global__ void k_convert_x(const float* __restrict__ x, _Float16* __restrict__ xh, int n) {
  int i = blockIdx.x * blockDim.x + threadIdx.x;
  if (i < n) xh[i] = (_Float16)x[i];
}

// Wq/Wk/Wv [H,D,DR] f32 -> Wt [3,H,DR,D] f16 (transposed: K contiguous for B-frags)
__global__ void k_trans_qkv(const float* __restrict__ Wq, const float* __restrict__ Wk,
                            const float* __restrict__ Wv, _Float16* __restrict__ Wt) {
  int i = blockIdx.x * blockDim.x + threadIdx.x;
  const int per = H_ * D_ * DR_;
  if (i >= 3 * per) return;
  int p = i / per, r = i % per;
  int h = r / (D_ * DR_);
  int r2 = r % (D_ * DR_);
  int d = r2 / DR_, e = r2 % DR_;
  const float* W = (p == 0) ? Wq : ((p == 1) ? Wk : Wv);
  Wt[(((size_t)p * H_ + h) * DR_ + e) * D_ + d] = (_Float16)W[((size_t)h * D_ + d) * DR_ + e];
}

// Wo [H*DR, D] f32 -> Wot [D, H*DR] f16
__global__ void k_trans_wo(const float* __restrict__ Wo, _Float16* __restrict__ Wot) {
  int i = blockIdx.x * blockDim.x + threadIdx.x;
  if (i >= D_ * (H_ * DR_)) return;
  int n = i / (H_ * DR_), k = i % (H_ * DR_);
  Wot[(size_t)n * (H_ * DR_) + k] = (_Float16)Wo[(size_t)k * D_ + n];
}

// ---- QKV projection: one wave = 32 rows x DR cols, z selects Q/K/V -----------
// Fully unrolled k-loop: no loop-carried fragment registers -> no copies, no
// WMMA WAR hazard NOPs; the scheduler pipelines loads across steps freely.
// Q is pre-scaled by 1/sqrt(DR) so flash attention skips the score scaling.

__global__ void __launch_bounds__(128)
k_qkv(const _Float16* __restrict__ xh, const _Float16* __restrict__ Wt,
      _Float16* __restrict__ Q, _Float16* __restrict__ K, _Float16* __restrict__ Vt) {
  int wave = threadIdx.x >> 5, lane = threadIdx.x & 31;
  int b = blockIdx.y / H_, h = blockIdx.y % H_;
  int proj = blockIdx.z;
  int row0 = blockIdx.x * 128 + wave * 32;

  const _Float16* A = xh + ((size_t)b * S_ + row0) * D_;
  const _Float16* W = Wt + (((size_t)proj * H_ + h) * DR_) * D_;

  v8f acc[2][4] = {};
#pragma unroll
  for (int k0 = 0; k0 < D_; k0 += 32) {
    v16h a0 = load_afrag(A + k0, D_);
    v16h a1 = load_afrag(A + (size_t)16 * D_ + k0, D_);
#pragma unroll
    for (int t = 0; t < 4; ++t) {
      v16h bf = load_bfrag(W + (size_t)(t * 16) * D_ + k0, D_);
      acc[0][t] = wmma_f16(a0, bf, acc[0][t]);
      acc[1][t] = wmma_f16(a1, bf, acc[1][t]);
    }
  }

  float osc = (proj == 0) ? 0.125f : 1.0f;   // fold 1/sqrt(64) into Q
  int n = lane & 15, hi = (lane >> 4) << 3;
  if (proj == 2) {  // V stored transposed: [B,H,DR,S]
    _Float16* dst = Vt + (((size_t)b * H_ + h) * DR_) * S_;
#pragma unroll
    for (int rb = 0; rb < 2; ++rb)
#pragma unroll
      for (int t = 0; t < 4; ++t)
#pragma unroll
        for (int r = 0; r < 8; ++r)
          dst[(size_t)(t * 16 + n) * S_ + row0 + rb * 16 + r + hi] = (_Float16)acc[rb][t][r];
  } else {          // Q/K stored [B,H,S,DR]
    _Float16* dst = ((proj == 0) ? Q : K) + (((size_t)b * H_ + h) * S_ + row0) * DR_;
#pragma unroll
    for (int rb = 0; rb < 2; ++rb)
#pragma unroll
      for (int t = 0; t < 4; ++t)
#pragma unroll
        for (int r = 0; r < 8; ++r)
          dst[(size_t)(rb * 16 + r + hi) * DR_ + t * 16 + n] = (_Float16)(acc[rb][t][r] * osc);
  }
}

// ---- Flash attention: one wave = 32 queries, key loop unrolled by 2 ----------
// Two distinct K-fragment buffers alternate roles each 32-key step: each buffer
// has a full step of prefetch distance and is only redefined, never copied.

__global__ void __launch_bounds__(128)
k_flash(const _Float16* __restrict__ Q, const _Float16* __restrict__ K,
        const _Float16* __restrict__ Vt, _Float16* __restrict__ attn) {
  __shared__ _Float16 pbuf[4][32][32];   // per-wave 32x32 P transpose tile
  int wave = threadIdx.x >> 5, lane = threadIdx.x & 31;
  int b = blockIdx.y / H_, h = blockIdx.y % H_;
  int row0 = blockIdx.x * 128 + wave * 32;

  const _Float16* Qp = Q  + (((size_t)b * H_ + h) * S_ + row0) * DR_;
  const _Float16* Kp = K  + (((size_t)b * H_ + h) * S_) * DR_;
  const _Float16* Vp = Vt + (((size_t)b * H_ + h) * DR_) * S_;

  v16h qa[2][2];
#pragma unroll
  for (int qb = 0; qb < 2; ++qb) {
    qa[qb][0] = load_afrag(Qp + (size_t)qb * 16 * DR_, DR_);
    qa[qb][1] = load_afrag(Qp + (size_t)qb * 16 * DR_ + 32, DR_);
  }

  v8f o[2][4] = {};
  float mrow[2][8], lrow[2][8];
#pragma unroll
  for (int qb = 0; qb < 2; ++qb)
#pragma unroll
    for (int r = 0; r < 8; ++r) { mrow[qb][r] = -1e30f; lrow[qb][r] = 0.0f; }

  _Float16* pb = &pbuf[wave][0][0];
  int n = lane & 15, hi = (lane >> 4) << 3;

  // one 32-key step: scores with kbc, prefetch kbn for key 'knext', softmax, PV
  auto step = [&](v16h (&kbc)[4], v16h (&kbn)[4], int key0, int knext) {
    v8f s[2][2];
#pragma unroll
    for (int qb = 0; qb < 2; ++qb) {
      v8f t0 = {};
      t0 = wmma_f16(qa[qb][0], kbc[0], t0);
      t0 = wmma_f16(qa[qb][1], kbc[1], t0);
      s[qb][0] = t0;
      v8f t1 = {};
      t1 = wmma_f16(qa[qb][0], kbc[2], t1);
      t1 = wmma_f16(qa[qb][1], kbc[3], t1);
      s[qb][1] = t1;
    }

    // prefetch next step's K fragments into the other buffer
    int kn = knext & (S_ - 1);
    kbn[0] = load_bfrag(Kp + (size_t)kn * DR_, DR_);
    kbn[1] = load_bfrag(Kp + (size_t)kn * DR_ + 32, DR_);
    kbn[2] = load_bfrag(Kp + (size_t)(kn + 16) * DR_, DR_);
    kbn[3] = load_bfrag(Kp + (size_t)(kn + 16) * DR_ + 32, DR_);

    // current block's V fragments, issued before the softmax VALU chain
    v16h vb[4];
#pragma unroll
    for (int t = 0; t < 4; ++t)
      vb[t] = load_bfrag(Vp + (size_t)(t * 16) * S_ + key0, S_);

    // online softmax; row stats replicated across the 16 lanes of each half
#pragma unroll
    for (int qb = 0; qb < 2; ++qb) {
      float p0[8], p1[8], alpha[8];
#pragma unroll
      for (int r = 0; r < 8; ++r) {
        float a = s[qb][0][r], c = s[qb][1][r];
        float mx = fmaxf(a, c);
#pragma unroll
        for (int msk = 8; msk >= 1; msk >>= 1)
          mx = fmaxf(mx, __shfl_xor(mx, msk, 16));
        float mnew = fmaxf(mrow[qb][r], mx);
        alpha[r] = __expf(mrow[qb][r] - mnew);
        p0[r] = __expf(a - mnew);
        p1[r] = __expf(c - mnew);
        float rs = p0[r] + p1[r];
#pragma unroll
        for (int msk = 8; msk >= 1; msk >>= 1)
          rs += __shfl_xor(rs, msk, 16);
        lrow[qb][r] = lrow[qb][r] * alpha[r] + rs;
        mrow[qb][r] = mnew;
      }
#pragma unroll
      for (int t = 0; t < 4; ++t)
#pragma unroll
        for (int r = 0; r < 8; ++r)
          o[qb][t][r] *= alpha[r];
      // stage P sub-tile (C-layout -> LDS row-major)
#pragma unroll
      for (int r = 0; r < 8; ++r) {
        pb[(qb * 16 + r + hi) * 32 + n]      = (_Float16)p0[r];
        pb[(qb * 16 + r + hi) * 32 + 16 + n] = (_Float16)p1[r];
      }
    }

    // per-wave transpose only: same-wave DS ops are in-order; wait for data return
    asm volatile("s_wait_dscnt 0" ::: "memory");
    v16h pa0 = load_afrag(pb, 32);
    v16h pa1 = load_afrag(pb + 16 * 32, 32);

    // O += P(16x32) @ V(32x64)
#pragma unroll
    for (int t = 0; t < 4; ++t) {
      o[0][t] = wmma_f16(pa0, vb[t], o[0][t]);
      o[1][t] = wmma_f16(pa1, vb[t], o[1][t]);
    }
  };

  v16h kb0[4], kb1[4];
  kb0[0] = load_bfrag(Kp, DR_);
  kb0[1] = load_bfrag(Kp + 32, DR_);
  kb0[2] = load_bfrag(Kp + (size_t)16 * DR_, DR_);
  kb0[3] = load_bfrag(Kp + (size_t)16 * DR_ + 32, DR_);

  for (int key0 = 0; key0 < S_; key0 += 64) {
    step(kb0, kb1, key0, key0 + 32);
    step(kb1, kb0, key0 + 32, key0 + 64);
  }

  // normalize and write head-concatenated output: attn[B,S,H*DR]
  _Float16* dst = attn + ((size_t)b * S_ + row0) * (H_ * DR_) + h * DR_;
#pragma unroll
  for (int qb = 0; qb < 2; ++qb) {
    float linv[8];
#pragma unroll
    for (int r = 0; r < 8; ++r) linv[r] = 1.0f / lrow[qb][r];
#pragma unroll
    for (int t = 0; t < 4; ++t)
#pragma unroll
      for (int r = 0; r < 8; ++r)
        dst[(size_t)(qb * 16 + r + hi) * (H_ * DR_) + t * 16 + n] =
            (_Float16)(o[qb][t][r] * linv[r]);
  }
}

// ---- Output projection: [B*S,512] @ Wo + bo, f32 out (fully unrolled) --------

__global__ void __launch_bounds__(128)
k_outproj(const _Float16* __restrict__ attn, const _Float16* __restrict__ Wot,
          const float* __restrict__ bo, float* __restrict__ out) {
  int wave = threadIdx.x >> 5, lane = threadIdx.x & 31;
  size_t row0 = (size_t)blockIdx.x * 128 + (size_t)wave * 32;
  int n0 = blockIdx.y * 64;
  const int KD = H_ * DR_;               // 512

  const _Float16* A = attn + row0 * KD;
  const _Float16* W = Wot + (size_t)n0 * KD;

  v8f acc[2][4] = {};
#pragma unroll
  for (int k0 = 0; k0 < KD; k0 += 32) {
    v16h a0 = load_afrag(A + k0, KD);
    v16h a1 = load_afrag(A + (size_t)16 * KD + k0, KD);
#pragma unroll
    for (int t = 0; t < 4; ++t) {
      v16h bf = load_bfrag(W + (size_t)(t * 16) * KD + k0, KD);
      acc[0][t] = wmma_f16(a0, bf, acc[0][t]);
      acc[1][t] = wmma_f16(a1, bf, acc[1][t]);
    }
  }

  int n = lane & 15, hi = (lane >> 4) << 3;
#pragma unroll
  for (int rb = 0; rb < 2; ++rb)
#pragma unroll
    for (int t = 0; t < 4; ++t)
#pragma unroll
      for (int r = 0; r < 8; ++r)
        out[(row0 + rb * 16 + r + hi) * D_ + n0 + t * 16 + n] =
            acc[rb][t][r] + bo[n0 + t * 16 + n];
}

// ---- Host launch -------------------------------------------------------------

extern "C" void kernel_launch(void* const* d_in, const int* in_sizes, int n_in,
                              void* d_out, int out_size, void* d_ws, size_t ws_size,
                              hipStream_t stream) {
  const float* x  = (const float*)d_in[0];
  const float* Wq = (const float*)d_in[1];
  const float* Wk = (const float*)d_in[2];
  const float* Wv = (const float*)d_in[3];
  const float* Wo = (const float*)d_in[4];
  const float* bo = (const float*)d_in[5];
  float* out = (float*)d_out;

  char* ws = (char*)d_ws;
  size_t off = 0;
  auto alloc = [&](size_t bytes) -> char* {
    char* p = ws + off;
    off += (bytes + 255) & ~(size_t)255;
    return p;
  };
  _Float16* xh  = (_Float16*)alloc((size_t)B_ * S_ * D_ * 2);        // x in f16
  _Float16* Wt  = (_Float16*)alloc((size_t)3 * H_ * DR_ * D_ * 2);   // Wq/Wk/Wv^T f16
  _Float16* Wot = (_Float16*)alloc((size_t)D_ * H_ * DR_ * 2);       // Wo^T f16
  _Float16* Qh  = (_Float16*)alloc((size_t)B_ * H_ * S_ * DR_ * 2);  // Q f16 (pre-scaled)
  _Float16* Kh  = (_Float16*)alloc((size_t)B_ * H_ * S_ * DR_ * 2);  // K f16
  _Float16* Vth = (_Float16*)alloc((size_t)B_ * H_ * DR_ * S_ * 2);  // V^T f16
  _Float16* At  = (_Float16*)alloc((size_t)B_ * S_ * H_ * DR_ * 2);  // concat attn out f16

  { int nx = B_ * S_ * D_;
    k_convert_x<<<(nx + 255) / 256, 256, 0, stream>>>(x, xh, nx); }
  { int nw = 3 * H_ * D_ * DR_;
    k_trans_qkv<<<(nw + 255) / 256, 256, 0, stream>>>(Wq, Wk, Wv, Wt); }
  { int nw = D_ * H_ * DR_;
    k_trans_wo<<<(nw + 255) / 256, 256, 0, stream>>>(Wo, Wot); }

  k_qkv<<<dim3(S_ / 128, B_ * H_, 3), 128, 0, stream>>>(xh, Wt, Qh, Kh, Vth);
  k_flash<<<dim3(S_ / 128, B_ * H_), 128, 0, stream>>>(Qh, Kh, Vth, At);
  k_outproj<<<dim3((B_ * S_) / 128, D_ / 64), 128, 0, stream>>>(At, Wot, bo, out);
}
